// PercolationM_31885837205969
// MI455X (gfx1250) — compile-verified, matrix operand order checked by reading, not run on hardware
//
#include <hip/hip_runtime.h>

// ---------------------------------------------------------------------------
// PercolationM on MI455X (gfx1250):
//  Kernel 1: one 256-thread block per 32x32 image. Async-copy image to LDS,
//            LDS-atomic union-find CCL, component-size histogram in LDS,
//            max(component size, background count) -> int per image (d_ws).
//  Kernel 2: per-wave WMMA ones-matmul sums 64 patch counts per output group
//            (exact in f16/f32), then >>6 for the integer mean, cast to f32.
// ---------------------------------------------------------------------------

typedef __attribute__((ext_vector_type(16))) _Float16 v16h;
typedef __attribute__((ext_vector_type(8)))  float    v8f;
typedef __attribute__((ext_vector_type(4)))  int      v4i;

typedef __attribute__((address_space(1))) v4i* gv4i_p;  // global int4*
typedef __attribute__((address_space(3))) v4i* lv4i_p;  // LDS int4*

#define IMG_PIX   1024   // 32*32
#define CCL_THREADS 256

#if __has_builtin(__builtin_amdgcn_global_load_async_to_lds_b128)
#define USE_ASYNC_LDS 1
#else
#define USE_ASYNC_LDS 0
#endif

__device__ __forceinline__ void wait_async0() {
#if __has_builtin(__builtin_amdgcn_s_wait_asynccnt)
    __builtin_amdgcn_s_wait_asynccnt(0);
#else
    asm volatile("s_wait_asynccnt 0" ::: "memory");
#endif
}

// Union-find: labels only decrease (atomicMin), chains strictly descend.
__device__ __forceinline__ int find_root(volatile int* L, int a) {
    int p = L[a];
    while (p != a) { a = p; p = L[a]; }
    return a;
}

__device__ __forceinline__ void merge_labels(int* L, int a, int b) {
    a = find_root((volatile int*)L, a);
    b = find_root((volatile int*)L, b);
    while (a != b) {
        if (b < a) { int t = a; a = b; b = t; }   // ensure a < b
        int old = atomicMin(&L[b], a);            // ds_min_rtn on LDS
        if (old == b) break;
        b = old;
    }
}

__global__ void __launch_bounds__(CCL_THREADS)
ccl_kernel(const float* __restrict__ x, int* __restrict__ out_counts) {
    __shared__ __align__(16) unsigned s_buf[IMG_PIX]; // image bits, then counts
    __shared__ int s_lbl[IMG_PIX];
    __shared__ int s_bg, s_max;

    const int img = blockIdx.x;
    const int tid = threadIdx.x;
    const float* gimg = x + (size_t)img * IMG_PIX;

    if (tid == 0) { s_bg = 0; s_max = 0; }

    // ---- stage 4KB image into LDS (async global->LDS data path) ----
#if USE_ASYNC_LDS
    __builtin_amdgcn_global_load_async_to_lds_b128(
        (gv4i_p)(gimg + tid * 4),
        (lv4i_p)(s_buf + tid * 4),
        0, 0);
    wait_async0();
#else
    ((uint4*)s_buf)[tid] = ((const uint4*)gimg)[tid];
#endif
    __syncthreads();

    // ---- init labels; count background pixels ----
    int zc = 0;
#pragma unroll
    for (int k = 0; k < 4; ++k) {
        int p = tid + k * CCL_THREADS;
        bool fg = (s_buf[p] != 0u);               // 0.0f has zero bit pattern
        s_lbl[p] = fg ? p : -1;
        zc += fg ? 0 : 1;
    }
    if (zc) atomicAdd(&s_bg, zc);
    __syncthreads();

    // ---- zero count array (reuse s_buf) + union right/down fg edges ----
#pragma unroll
    for (int k = 0; k < 4; ++k) s_buf[tid + k * CCL_THREADS] = 0u;

    int* L = s_lbl;
#pragma unroll
    for (int k = 0; k < 4; ++k) {
        int p = tid + k * CCL_THREADS;
        if (L[p] >= 0) {                          // fg flag is monotone-invariant
            int xc = p & 31;
            int yr = p >> 5;
            if (xc < 31 && L[p + 1]  >= 0) merge_labels(L, p, p + 1);
            if (yr < 31 && L[p + 32] >= 0) merge_labels(L, p, p + 32);
        }
    }
    __syncthreads();

    // ---- component-size histogram ----
#pragma unroll
    for (int k = 0; k < 4; ++k) {
        int p = tid + k * CCL_THREADS;
        if (L[p] >= 0) {
            int r = find_root((volatile int*)L, p);
            atomicAdd(&s_buf[r], 1u);             // ds_add_rtn on LDS
        }
    }
    __syncthreads();

    // ---- block max over counts, compare with background count ----
    int lm = 0;
#pragma unroll
    for (int k = 0; k < 4; ++k) {
        int c = (int)s_buf[tid + k * CCL_THREADS];
        lm = lm > c ? lm : c;
    }
    atomicMax(&s_max, lm);
    __syncthreads();

    if (tid == 0) {
        int best = s_max > s_bg ? s_max : s_bg;
        out_counts[img] = best;
    }
}

// ---------------------------------------------------------------------------
// Patch-mean via WMMA: D = A(16x64 counts, split K=2x32) x B(ones) + C.
// Counts <= 1024 are exact in f16; sums <= 65536 exact in f32.
// A layout (16-bit 16x32, wave32): lane m   -> M=m, elems 0..7 = K0..7,
//                                               elems 8..15 = K16..23
//                                  lane m+16-> M=m, elems 0..7 = K8..15,
//                                               elems 8..15 = K24..31
// D layout: VGPR r, lane 0 -> (M=r, N=0); lane 16 -> (M=8+r, N=0).
// ---------------------------------------------------------------------------
__global__ void __launch_bounds__(32)
reduce_kernel(const int* __restrict__ counts, float* __restrict__ out) {
    const int w    = blockIdx.x;       // 16 output groups per wave
    const int lane = threadIdx.x;      // 0..31, wave32
    const int m    = lane & 15;
    const int hi   = lane >> 4;
    const int g    = w * 16 + m;
    const int* c64 = counts + g * 64;

    const int kb0 = hi * 8;            // 0 or 8
    const int kb1 = 16 + hi * 8;       // 16 or 24

    v16h a0, a1, b;
#pragma unroll
    for (int i = 0; i < 8; ++i) {
        a0[i]     = (_Float16)c64[kb0 + i];
        a0[8 + i] = (_Float16)c64[kb1 + i];
        a1[i]     = (_Float16)c64[32 + kb0 + i];
        a1[8 + i] = (_Float16)c64[32 + kb1 + i];
    }
#pragma unroll
    for (int i = 0; i < 16; ++i) b[i] = (_Float16)1.0f;  // ones: layout-free

    v8f acc = {};
    acc = __builtin_amdgcn_wmma_f32_16x16x32_f16(false, a0, false, b,
                                                 (short)0, acc, false, false);
    acc = __builtin_amdgcn_wmma_f32_16x16x32_f16(false, a1, false, b,
                                                 (short)0, acc, false, false);

    if ((lane & 15) == 0) {            // N=0 column: lanes 0 and 16
#pragma unroll
        for (int r = 0; r < 8; ++r) {
            int s = (int)acc[r];       // exact integer sum of 64 counts
            out[w * 16 + hi * 8 + r] = (float)(s >> 6);  // // P with P=64
        }
    }
}

extern "C" void kernel_launch(void* const* d_in, const int* in_sizes, int n_in,
                              void* d_out, int out_size, void* d_ws, size_t ws_size,
                              hipStream_t stream) {
    (void)n_in; (void)out_size; (void)ws_size;
    const float* x = (const float*)d_in[0];
    float* out = (float*)d_out;

    const int n_img  = in_sizes[0] >> 10;  // /1024 = 30720 images
    int* counts = (int*)d_ws;              // n_img ints of scratch

    ccl_kernel<<<n_img, CCL_THREADS, 0, stream>>>(x, counts);

    const int groups = n_img >> 6;         // /64 patches = 480 outputs
    reduce_kernel<<<groups >> 4, 32, 0, stream>>>(counts, out);
}
